// MultiQueryAttention_81269371174914
// MI455X (gfx1250) — matrix operand order repaired and checked by reading, not compiled, not run on hardware
//
#include <hip/hip_runtime.h>

typedef __attribute__((ext_vector_type(16))) __bf16 v16bf;
typedef __attribute__((ext_vector_type(8)))  float  v8f;
typedef __attribute__((ext_vector_type(4)))  unsigned v4u;
typedef __attribute__((ext_vector_type(8)))  int      v8i;
typedef __attribute__((ext_vector_type(4)))  int      v4i;

struct alignas(16) U4 { unsigned x, y, z, w; };
union Frag { v16bf v; U4 q[2]; unsigned u[8]; };

__device__ __forceinline__ unsigned short f2bf(float f) {
  unsigned u = __float_as_uint(f);
  u += 0x7FFFu + ((u >> 16) & 1u);        // round-to-nearest-even
  return (unsigned short)(u >> 16);
}

__device__ __forceinline__ v8f wmma_bf16(const Frag& a, const Frag& b, v8f c) {
  return __builtin_amdgcn_wmma_f32_16x16x32_bf16(false, a.v, false, b.v,
                                                 (short)0, c, false, false);
}

// ---------------------------------------------------------------------------
// TDM: 2D tile load Global -> LDS (CDNA5 TENSOR_LOAD_TO_LDS).
// D# built per ISA 8.3-8.5:  group0 = {count/type, lds_addr, global_addr},
// group1 = {data_size, tensor_dim0/1, tile_dim0/1, dim0_stride}.
// dsz_enc: 0=1B,1=2B,2=4B.  tile0/tensor_dim0/stride0 in data_size units.
// This toolchain exposes the 6-arg builtin:
//   (uint32x4 g0, int32x8 g1, int32x4 g2, int32x4 g3, int32x8 g4, i32 cpol)
// ---------------------------------------------------------------------------
__device__ __forceinline__ void tdm_load_2d(unsigned lds_addr, const void* gptr,
                                            unsigned dsz_enc, unsigned tile0,
                                            unsigned tile1, unsigned tdim0,
                                            unsigned stride0) {
  unsigned long long ga = (unsigned long long)gptr;
  v4u g0;
  g0.x = 1u;                                        // count=1 (valid), user mode
  g0.y = lds_addr;                                  // LDS byte address
  g0.z = (unsigned)ga;                              // global_addr[31:0]
  g0.w = (unsigned)((ga >> 32) & 0x01FFFFFFu)       // global_addr[56:32]
         | 0x80000000u;                             // type=2 ("image")
  int w0 = (int)(dsz_enc << 16);                    // mask=0, data_size
  int w1 = (int)((tdim0 & 0xFFFFu) << 16);          // tensor_dim0[15:0]
  int w2 = (int)((tdim0 >> 16) & 0xFFFFu);          // tensor_dim0[31:16], dim1 lo=0
  int w3 = (int)(0x10u | (tile0 << 16));            // tensor_dim1=1M rows, tile_dim0
  int w4 = (int)(tile1 & 0xFFFFu);                  // tile_dim1, tile_dim2=0
  int w5 = (int)stride0;                            // tensor_dim0_stride[31:0]
  v8i g1 = { w0, w1, w2, w3, w4, w5, 0, 0 };
  v4i g2 = { 1, 1, 0, 0 };                          // tensor_dim2/3 benign
  v4i g3 = { 0, 0, 0, 0 };
  v8i g4 = { 0, 0, 0, 0, 0, 0, 0, 0 };
  __builtin_amdgcn_tensor_load_to_lds(g0, g1, g2, g3, g4, 0);
}

// ---------------------------------------------------------------------------
// f32 -> bf16 elementwise convert (activations)
// ---------------------------------------------------------------------------
__global__ void cvt_f32_bf16(const float* __restrict__ in,
                             unsigned short* __restrict__ out, int n) {
  int i = blockIdx.x * 256 + threadIdx.x;
  if (i < n) out[i] = f2bf(in[i]);
}

// Weights: f32 (K x N) -> bf16 pair-packed Bp (N x K/2 dwords):
// Bp[n*K/2 + kp] = {W[2kp+1][n], W[2kp][n]}   (done ONCE, reused by all tiles)
__global__ void pack_w_bf16(const float* __restrict__ W, unsigned* __restrict__ Bp,
                            int K, int N) {
  int K2 = K >> 1;
  int i = blockIdx.x * 256 + threadIdx.x;     // over N*K2
  if (i >= N * K2) return;
  int n = i / K2, kp = i - n * K2;
  unsigned lo = f2bf(W[(size_t)(2 * kp) * N + n]);
  unsigned hi = f2bf(W[(size_t)(2 * kp + 1) * N + n]);
  Bp[i] = lo | (hi << 16);
}

// V (B*S,128) -> VT (B,128,S)
__global__ void transpose_v(const unsigned short* __restrict__ V,
                            unsigned short* __restrict__ VT) {
  int i = blockIdx.x * 256 + threadIdx.x;   // over B*S*128
  int d = i & 127;
  int s = (i >> 7) & 2047;
  int b = i >> 18;
  VT[((size_t)b * 128 + d) * 2048 + s] = V[i];
}

// ---------------------------------------------------------------------------
// C = A(MxK bf16 row-major) @ B(KxN, given pair-packed Bp N x K/2) + bias.
// Block tile 128x64, BK=32, 8 waves; wave = 32x32 via 2x2 WMMA.
// TDM double-buffered staging: wave0 issues TENSOR_LOAD_TO_LDS for tile t+1,
// waits tensorcnt<=2 (pair for tile t complete, in-order), block barriers,
// all waves compute WMMAs from LDS buffer t.
// ---------------------------------------------------------------------------
__global__ __launch_bounds__(256) void gemm_bf16_wmma(
    const unsigned short* __restrict__ A, const unsigned* __restrict__ Bp,
    const float* __restrict__ bias, unsigned short* __restrict__ Cb,
    float* __restrict__ Cf, int M, int N, int K) {
  __shared__ __align__(16) unsigned short sA[2][128 * 32]; // 2 x 8KB
  __shared__ __align__(16) unsigned       sB[2][64 * 16];  // 2 x 4KB
  const int t  = threadIdx.x;
  const int w  = t >> 5, lane = t & 31;
  const int hf = lane >> 4, ln = lane & 15;
  const int wm = w & 3, wn = w >> 2;
  const int m0 = blockIdx.y * 128, n0 = blockIdx.x * 64;
  const int K2 = K >> 1;
  const int T  = K >> 5;

  v8f acc00 = {}, acc01 = {}, acc10 = {}, acc11 = {};

  if (w == 0) {
    tdm_load_2d((unsigned)(unsigned long long)&sA[0][0],
                A + (size_t)m0 * K, 1u, 32u, 128u, (unsigned)K, (unsigned)K);
    tdm_load_2d((unsigned)(unsigned long long)&sB[0][0],
                Bp + (size_t)n0 * K2, 2u, 16u, 64u, (unsigned)K2, (unsigned)K2);
  }

  for (int kt = 0; kt < T; ++kt) {
    const int cur = kt & 1;
    if (w == 0) {
      if (kt + 1 < T) {
        const int k0n = (kt + 1) * 32;
        tdm_load_2d((unsigned)(unsigned long long)&sA[1 - cur][0],
                    A + (size_t)m0 * K + k0n, 1u, 32u, 128u,
                    (unsigned)K, (unsigned)K);
        tdm_load_2d((unsigned)(unsigned long long)&sB[1 - cur][0],
                    Bp + (size_t)n0 * K2 + (k0n >> 1), 2u, 16u, 64u,
                    (unsigned)K2, (unsigned)K2);
        __builtin_amdgcn_s_wait_tensorcnt(2);   // current pair complete
      } else {
        __builtin_amdgcn_s_wait_tensorcnt(0);
      }
    }
    __syncthreads();                            // LDS[cur] visible to all waves

    const U4* sA4 = (const U4*)&sA[cur][0];
    const U4* sB4 = (const U4*)&sB[cur][0];
    Frag a0, a1, b0, b1;
    int ra = wm * 32 + ln;
    a0.q[0] = sA4[ra * 4 + hf * 2];        a0.q[1] = sA4[ra * 4 + hf * 2 + 1];
    a1.q[0] = sA4[(ra + 16) * 4 + hf * 2]; a1.q[1] = sA4[(ra + 16) * 4 + hf * 2 + 1];
    int nb = wn * 32 + ln;
    b0.q[0] = sB4[nb * 4 + hf * 2];        b0.q[1] = sB4[nb * 4 + hf * 2 + 1];
    b1.q[0] = sB4[(nb + 16) * 4 + hf * 2]; b1.q[1] = sB4[(nb + 16) * 4 + hf * 2 + 1];

    acc00 = wmma_bf16(a0, b0, acc00);
    acc01 = wmma_bf16(a0, b1, acc01);
    acc10 = wmma_bf16(a1, b0, acc10);
    acc11 = wmma_bf16(a1, b1, acc11);
    __syncthreads();                            // done reading before next DMA
  }

  #pragma unroll
  for (int i = 0; i < 2; ++i) {
    #pragma unroll
    for (int j = 0; j < 2; ++j) {
      v8f acc = (i == 0) ? (j == 0 ? acc00 : acc01) : (j == 0 ? acc10 : acc11);
      int col = n0 + wn * 32 + j * 16 + ln;
      float bb = bias ? bias[col] : 0.0f;
      #pragma unroll
      for (int vv = 0; vv < 8; ++vv) {
        int row = m0 + wm * 32 + i * 16 + vv + 8 * hf;
        float val = acc[vv] + bb;
        if (Cf) Cf[(size_t)row * N + col] = val;
        else    Cb[(size_t)row * N + col] = f2bf(val);
      }
    }
  }
}

// ---------------------------------------------------------------------------
// Flash-style multi-query attention.
// Qb: (B*S, 2048) bf16, Kb: (B*S, 128) bf16, VT: (B, 128, S) bf16,
// Ob: (B*S, 2048) bf16.  Grid = B*H*(S/128), 8 waves, 16 query rows per wave.
// ---------------------------------------------------------------------------
__global__ __launch_bounds__(256) void mqa_attn(
    const unsigned short* __restrict__ Qb, const unsigned short* __restrict__ Kb,
    const unsigned short* __restrict__ VT, unsigned short* __restrict__ Ob) {
  constexpr int S = 2048, D = 128;
  constexpr float SCALE = 0.0883883476483184f;   // 1/sqrt(128)
  __shared__ __align__(16) unsigned lp[2048];    // 8 waves * 16x32 bf16 patch

  const int t  = threadIdx.x;
  const int w  = t >> 5, lane = t & 31;
  const int hf = lane >> 4, ln = lane & 15;

  int bid = blockIdx.x;
  int qt  = bid & 15;          // S/128 = 16
  int h   = (bid >> 4) & 15;   // H = 16
  int b   = bid >> 8;          // B = 2

  const int q0 = qt * 128 + w * 16;

  Frag aq[4];
  const U4* qp = (const U4*)(Qb + (size_t)(b * S + q0 + ln) * 2048 + h * D);
  #pragma unroll
  for (int c = 0; c < 4; ++c) {
    aq[c].q[0] = qp[c * 4 + hf * 2];
    aq[c].q[1] = qp[c * 4 + hf * 2 + 1];
  }

  v8f o[8] = {};
  float mr[8], lr[8];
  #pragma unroll
  for (int vv = 0; vv < 8; ++vv) { mr[vv] = -1e30f; lr[vv] = 0.0f; }

  unsigned short* sp = (unsigned short*)(lp + w * 256);
  const U4*      sp4 = (const U4*)(lp + w * 256);

  const int ntile = (q0 + 15) / 32 + 1;           // causal limit
  for (int kt = 0; kt < ntile; ++kt) {
    const int k0 = kt * 32;

    v8f s0 = {}, s1 = {};
    #pragma unroll
    for (int c = 0; c < 4; ++c) {
      Frag kb0, kb1;
      const U4* kp0 = (const U4*)(Kb + (size_t)(b * S + k0 + ln) * D + c * 32);
      const U4* kp1 = (const U4*)(Kb + (size_t)(b * S + k0 + 16 + ln) * D + c * 32);
      kb0.q[0] = kp0[hf * 2]; kb0.q[1] = kp0[hf * 2 + 1];
      kb1.q[0] = kp1[hf * 2]; kb1.q[1] = kp1[hf * 2 + 1];
      s0 = wmma_bf16(aq[c], kb0, s0);
      s1 = wmma_bf16(aq[c], kb1, s1);
    }

    float p0[8], p1[8], alpha[8];
    #pragma unroll
    for (int vv = 0; vv < 8; ++vv) {
      int qi = q0 + vv + 8 * hf;
      float x0 = s0[vv] * SCALE;
      float x1 = s1[vv] * SCALE;
      if (k0 + ln > qi)      x0 = -1e30f;
      if (k0 + 16 + ln > qi) x1 = -1e30f;
      float rmx = fmaxf(x0, x1);
      #pragma unroll
      for (int mk = 1; mk < 16; mk <<= 1) rmx = fmaxf(rmx, __shfl_xor(rmx, mk, 32));
      float mn = fmaxf(mr[vv], rmx);
      alpha[vv] = __expf(mr[vv] - mn);
      mr[vv] = mn;
      p0[vv] = __expf(x0 - mn);
      p1[vv] = __expf(x1 - mn);
      float rs = p0[vv] + p1[vv];
      #pragma unroll
      for (int mk = 1; mk < 16; mk <<= 1) rs += __shfl_xor(rs, mk, 32);
      lr[vv] = lr[vv] * alpha[vv] + rs;
    }
    #pragma unroll
    for (int j = 0; j < 8; ++j)
      #pragma unroll
      for (int vv = 0; vv < 8; ++vv) o[j][vv] *= alpha[vv];

    // P (C-layout f32) -> bf16 A-fragment via per-wave LDS patch
    #pragma unroll
    for (int vv = 0; vv < 8; ++vv) {
      int m = vv + 8 * hf;
      sp[m * 32 + ln]      = f2bf(p0[vv]);
      sp[m * 32 + 16 + ln] = f2bf(p1[vv]);
    }
    asm volatile("s_wait_dscnt 0" ::: "memory");   // wave-local LDS RAW
    Frag ap;
    ap.q[0] = sp4[ln * 4 + hf * 2];
    ap.q[1] = sp4[ln * 4 + hf * 2 + 1];

    #pragma unroll
    for (int j = 0; j < 8; ++j) {
      Frag bvf;
      const U4* vp = (const U4*)(VT + ((size_t)b * D + j * 16 + ln) * S + k0);
      bvf.q[0] = vp[hf * 2];
      bvf.q[1] = vp[hf * 2 + 1];
      o[j] = wmma_bf16(ap, bvf, o[j]);
    }
  }

  float linv[8];
  #pragma unroll
  for (int vv = 0; vv < 8; ++vv) linv[vv] = 1.0f / lr[vv];
  #pragma unroll
  for (int j = 0; j < 8; ++j) {
    #pragma unroll
    for (int vv = 0; vv < 8; ++vv) {
      int m = vv + 8 * hf;
      Ob[(size_t)(b * S + q0 + m) * 2048 + h * D + j * 16 + ln] =
          f2bf(o[j][vv] * linv[vv]);
    }
  }
}

// ---------------------------------------------------------------------------
extern "C" void kernel_launch(void* const* d_in, const int* in_sizes, int n_in,
                              void* d_out, int out_size, void* d_ws, size_t ws_size,
                              hipStream_t stream) {
  (void)in_sizes; (void)n_in; (void)out_size; (void)ws_size;
  const float* x  = (const float*)d_in[0];
  // d_in[1] = mask (causal, handled analytically)
  const float* Wq = (const float*)d_in[2];
  const float* bq = (const float*)d_in[3];
  const float* Wk = (const float*)d_in[4];
  const float* bk = (const float*)d_in[5];
  const float* Wv = (const float*)d_in[6];
  const float* bv = (const float*)d_in[7];
  const float* Wo = (const float*)d_in[8];
  const float* bo = (const float*)d_in[9];

  constexpr int B = 2, S = 2048, HID = 2048, D = 128, H = 16;
  constexpr int M = B * S;                 // 4096

  char* base = (char*)d_ws;
  size_t off = 0;
  auto take = [&](size_t bytes) {
    void* p = base + off;
    off += (bytes + 255) & ~(size_t)255;
    return p;
  };
  unsigned short* xb  = (unsigned short*)take((size_t)M * HID * 2);
  unsigned*       Wqp = (unsigned*)take((size_t)HID * (HID / 2) * 4);
  unsigned*       Wkp = (unsigned*)take((size_t)D * (HID / 2) * 4);
  unsigned*       Wvp = (unsigned*)take((size_t)D * (HID / 2) * 4);
  unsigned*       Wop = (unsigned*)take((size_t)HID * (HID / 2) * 4);
  unsigned short* Qb  = (unsigned short*)take((size_t)M * HID * 2);
  unsigned short* Kbf = (unsigned short*)take((size_t)M * D * 2);
  unsigned short* Vbf = (unsigned short*)take((size_t)M * D * 2);
  unsigned short* VTb = (unsigned short*)take((size_t)M * D * 2);
  unsigned short* Ob  = (unsigned short*)take((size_t)M * HID * 2);

  const int nx = M * HID;
  cvt_f32_bf16<<<(nx + 255) / 256, 256, 0, stream>>>(x, xb, nx);
  const int nwq = HID * (HID / 2), nwk = D * (HID / 2);
  pack_w_bf16<<<(nwq + 255) / 256, 256, 0, stream>>>(Wq, Wqp, HID, HID);
  pack_w_bf16<<<(nwk + 255) / 256, 256, 0, stream>>>(Wk, Wkp, HID, D);
  pack_w_bf16<<<(nwk + 255) / 256, 256, 0, stream>>>(Wv, Wvp, HID, D);
  pack_w_bf16<<<(nwq + 255) / 256, 256, 0, stream>>>(Wo, Wop, HID, HID);

  // Q = x @ Wq + bq          (4096 x 2048 x 2048)
  gemm_bf16_wmma<<<dim3(HID / 64, M / 128), 256, 0, stream>>>(
      xb, Wqp, bq, Qb, nullptr, M, HID, HID);
  // K = x @ Wk + bk          (4096 x 128 x 2048)
  gemm_bf16_wmma<<<dim3(D / 64, M / 128), 256, 0, stream>>>(
      xb, Wkp, bk, Kbf, nullptr, M, D, HID);
  // V = x @ Wv + bv
  gemm_bf16_wmma<<<dim3(D / 64, M / 128), 256, 0, stream>>>(
      xb, Wvp, bv, Vbf, nullptr, M, D, HID);
  // V^T for contiguous PV B-fragments
  transpose_v<<<(M * D) / 256, 256, 0, stream>>>(Vbf, VTb);

  // attention: grid = B * H * (S/128) = 512
  mqa_attn<<<B * H * (S / 128), 256, 0, stream>>>(Qb, Kbf, VTb, Ob);

  // out = attOut @ Wo + bo -> f32
  gemm_bf16_wmma<<<dim3(HID / 64, M / 128), 256, 0, stream>>>(
      Ob, Wop, bo, nullptr, (float*)d_out, M, HID, HID);
}
